// ContrastiveLoss_17437567222535
// MI455X (gfx1250) — matrix operand order repaired and checked by reading, not compiled, not run on hardware
//
#include <hip/hip_runtime.h>
#include <hip/hip_bf16.h>

// ---------------------------------------------------------------------------
// Contrastive loss: sum_i [ logsumexp_j(X_i . Y_j / T) - (X_i . Y_i / T) ]
// N = 8192, C = 512, T = 0.07.
//
// MI455X strategy:
//  - bf16x3 split GEMM (xh.yh + xh.yl + xl.yh) on V_WMMA_F32_16X16X32_BF16:
//    8x FLOP/instr vs native f32 WMMA, ~fp32 accuracy, net ~2.7x win.
//  - Fused online logsumexp in registers: the 8192^2 score matrix never
//    touches memory (256 MB avoided).
//  - B panel (32 cols x 512 K, hi+lo = 64 KB) staged per workgroup via
//    GLOBAL_LOAD_ASYNC_TO_LDS_B128 (ASYNCcnt engine), shared by all 8 waves
//    (8x L2 traffic cut), and DOUBLE-BUFFERED (128 KB of the WGP's 320 KB
//    LDS): panel g+1 streams in the background while WMMAs consume panel g;
//    one s_wait_asynccnt + one barrier per panel, zero staging bubbles.
//  - exp/log in base-2 domain -> bare v_exp_f32 / v_log_f32 trans ops that
//    co-execute with 16-bit WMMA (tracked as TRANS, cdna5 flow/sync 8.4).
// ---------------------------------------------------------------------------

typedef __attribute__((ext_vector_type(16))) __bf16 v16bf;
typedef __attribute__((ext_vector_type(8)))  __bf16 v8bf;
typedef __attribute__((ext_vector_type(8)))  float  v8f;

#define N_ROWS 8192
#define C_DIM  512
#define WAVES_PER_BLOCK 8
#define ROWS_PER_WAVE 16
#define NT 2                    // 2 column tiles = 32 columns per LDS panel
#define PANEL_COLS (NT * 16)    // 32
#define PANEL_ELTS (PANEL_COLS * C_DIM)        // 16384 bf16 per matrix
#define N_PANELS (N_ROWS / PANEL_COLS)         // 256
#define SCALE2 (1.4426950408889634f / 0.07f)   // log2(e) / T
#define LN2 0.6931471805599453f

// fp32 -> (bf16 hi, bf16 residual)
__global__ void split_bf16_kernel(const float* __restrict__ x,
                                  __bf16* __restrict__ hi,
                                  __bf16* __restrict__ lo, int n) {
  int i = blockIdx.x * blockDim.x + threadIdx.x;
  if (i < n) {
    float f = x[i];
    __bf16 h = (__bf16)f;
    hi[i] = h;
    lo[i] = (__bf16)(f - (float)h);
  }
}

__global__ __launch_bounds__(WAVES_PER_BLOCK * 32)
void gemm_lse_kernel(const __bf16* __restrict__ Xh, const __bf16* __restrict__ Xl,
                     const __bf16* __restrict__ Yh, const __bf16* __restrict__ Yl,
                     float* __restrict__ out) {
  // Double-buffered B panel: per buffer, [0, PANEL_ELTS) = hi,
  // [PANEL_ELTS, 2*PANEL_ELTS) = lo. 2 x 64 KB = 128 KB (<= 320 KB/WGP).
  __shared__ __bf16 sB[2][2 * PANEL_ELTS];

  const int tid  = threadIdx.x;
  const int lane = tid & 31;
  const int wave = tid >> 5;
  const int bn   = lane & 15;   // A-row / B-col / D-col within tile
  const int rg   = lane >> 4;   // half-group: K slice select + D row half
  const int row_base = (blockIdx.x * WAVES_PER_BLOCK + wave) * ROWS_PER_WAVE;

  // WMMA 16-bit fragment k order used here (contiguous, same for A and B; not the ISA table order):
  //  A 16x32: lane holds row (lane&15); 16 contiguous K at 16*(lane>>4).
  //  B 32x16: lane holds col (lane&15); 16 contiguous K at 16*(lane>>4).
  const int koffA = rg * 16;
  const int koffB = rg * 16;

  const unsigned sB0_base = (unsigned)(uintptr_t)&sB[0][0];
  const unsigned sB1_base = (unsigned)(uintptr_t)&sB[1][0];

  float m_run[8], s_run[8], pos[8];   // slot j <-> row row_base + j + 8*rg
#pragma unroll
  for (int j = 0; j < 8; ++j) { m_run[j] = -3.0e38f; s_run[j] = 0.f; pos[j] = 0.f; }

  const __bf16* xh_row = Xh + (size_t)(row_base + bn) * C_DIM;
  const __bf16* xl_row = Xl + (size_t)(row_base + bn) * C_DIM;

  // ---- Async panel stage: 32 consecutive Y rows = contiguous 32 KB/matrix.
  auto stage_panel = [&](int g, unsigned buf_base) {
    const __bf16* gH = Yh + (size_t)g * PANEL_ELTS;
    const __bf16* gL = Yl + (size_t)g * PANEL_ELTS;
#pragma unroll
    for (int i = 0; i < (PANEL_ELTS * 2) / (256 * 16); ++i) {  // 8 chunks/thread
      const int c    = tid + i * 256;       // 16-byte chunk index
      const int voff = c * 16;              // byte offset into panel
      const unsigned ldsH = buf_base + (unsigned)voff;
      const unsigned ldsL = buf_base + (unsigned)(PANEL_ELTS * 2) + (unsigned)voff;
      asm volatile("global_load_async_to_lds_b128 %0, %1, %2"
                   :: "v"(ldsH), "v"(voff), "s"(gH) : "memory");
      asm volatile("global_load_async_to_lds_b128 %0, %1, %2"
                   :: "v"(ldsL), "v"(voff), "s"(gL) : "memory");
    }
  };

  // Prologue: stage panel 0 into buffer 0, publish to all waves.
  stage_panel(0, sB0_base);
  asm volatile("s_wait_asynccnt 0x0" ::: "memory");
  __syncthreads();

  for (int g = 0; g < N_PANELS; ++g) {
    const int  buf      = g & 1;
    const __bf16* sCur  = &sB[buf][0];

    // Kick off next panel into the other buffer; it streams via the async
    // engine while the WMMAs below consume the current buffer.
    if (g + 1 < N_PANELS)
      stage_panel(g + 1, buf ? sB0_base : sB1_base);

    const int col0 = g * PANEL_COLS;
    v8f acc[NT] = {};

#pragma unroll 2
    for (int kc = 0; kc < C_DIM / 32; ++kc) {
      const int k0 = kc * 32;
      // A fragments from global (16 rows x 1 KB x2, WGP$-resident)
      v8bf ah0 = *(const v8bf*)(xh_row + k0 + koffA);
      v8bf ah1 = *(const v8bf*)(xh_row + k0 + koffA + 8);
      v8bf al0 = *(const v8bf*)(xl_row + k0 + koffA);
      v8bf al1 = *(const v8bf*)(xl_row + k0 + koffA + 8);
      v16bf ah = __builtin_shufflevector(ah0, ah1, 0,1,2,3,4,5,6,7,8,9,10,11,12,13,14,15);
      v16bf al = __builtin_shufflevector(al0, al1, 0,1,2,3,4,5,6,7,8,9,10,11,12,13,14,15);

#pragma unroll
      for (int t = 0; t < NT; ++t) {
        const int cl = t * 16 + bn;                       // panel-local column
        const __bf16* bhp = &sCur[(size_t)cl * C_DIM + k0 + koffB];
        const __bf16* blp = bhp + PANEL_ELTS;
        v16bf bh = *(const v16bf*)bhp;                    // 2x ds_load_b128
        v16bf bl = *(const v16bf*)blp;
        acc[t] = __builtin_amdgcn_wmma_f32_16x16x32_bf16(
            false, ah, false, bh, (short)0, acc[t], false, false);
        acc[t] = __builtin_amdgcn_wmma_f32_16x16x32_bf16(
            false, ah, false, bl, (short)0, acc[t], false, false);
        acc[t] = __builtin_amdgcn_wmma_f32_16x16x32_bf16(
            false, al, false, bh, (short)0, acc[t], false, false);
      }
    }

    // ---- Online logsumexp update, base-2 domain (native v_exp_f32). ----
#pragma unroll
    for (int t = 0; t < NT; ++t) {
      const int col = col0 + t * 16 + bn;
#pragma unroll
      for (int j = 0; j < 8; ++j) {
        float v2  = acc[t][j] * SCALE2;
        int   row = row_base + j + rg * 8;
        pos[j] += (col == row) ? v2 : 0.0f;   // diagonal, select (no EXEC change)
        float mn  = fmaxf(m_run[j], v2);
        s_run[j]  = s_run[j] * __builtin_amdgcn_exp2f(m_run[j] - mn)
                  + __builtin_amdgcn_exp2f(v2 - mn);
        m_run[j]  = mn;
      }
    }

    // Publish next panel & retire current one: our async writes are done
    // (asynccnt), everyone's reads/writes are ordered by the barrier.
    asm volatile("s_wait_asynccnt 0x0" ::: "memory");
    __syncthreads();
  }

  // Merge (m, s, pos) across the 16 lanes of each half-group
  // (xor offsets 1..8 never cross bit 4, so row halves stay separate).
  float partial = 0.f;
#pragma unroll
  for (int j = 0; j < 8; ++j) {
    float m = m_run[j], s = s_run[j], p = pos[j];
#pragma unroll
    for (int off = 1; off < 16; off <<= 1) {
      float mo = __shfl_xor(m, off, 32);
      float so = __shfl_xor(s, off, 32);
      float po = __shfl_xor(p, off, 32);
      float mn = fmaxf(m, mo);
      s = s * __builtin_amdgcn_exp2f(m - mn) + so * __builtin_amdgcn_exp2f(mo - mn);
      m = mn;
      p += po;
    }
    partial += (m + __builtin_amdgcn_logf(s)) - p;   // log2 domain
  }
  // Lane 0: rows [row_base, +8); lane 16: rows [row_base+8, +16).
  if (bn == 0) atomicAdd(out, partial * LN2);
}

extern "C" void kernel_launch(void* const* d_in, const int* in_sizes, int n_in,
                              void* d_out, int out_size, void* d_ws, size_t ws_size,
                              hipStream_t stream) {
  const float* X = (const float*)d_in[0];
  const float* Y = (const float*)d_in[1];
  float* out = (float*)d_out;

  const size_t nelem = (size_t)N_ROWS * C_DIM;
  char* ws = (char*)d_ws;
  __bf16* Xh = (__bf16*)(ws);
  __bf16* Xl = (__bf16*)(ws + 2 * nelem);
  __bf16* Yh = (__bf16*)(ws + 4 * nelem);
  __bf16* Yl = (__bf16*)(ws + 6 * nelem);

  hipMemsetAsync(d_out, 0, sizeof(float), stream);

  const int n = (int)nelem;
  split_bf16_kernel<<<(n + 255) / 256, 256, 0, stream>>>(X, Xh, Xl, n);
  split_bf16_kernel<<<(n + 255) / 256, 256, 0, stream>>>(Y, Yh, Yl, n);

  gemm_lse_kernel<<<N_ROWS / (WAVES_PER_BLOCK * ROWS_PER_WAVE),
                    WAVES_PER_BLOCK * 32, 0, stream>>>(Xh, Xl, Yh, Yl, out);
}
